// TensorProductMemory_63024350101866
// MI455X (gfx1250) — matrix operand order repaired and checked by reading, not compiled, not run on hardware
//
#include <hip/hip_runtime.h>

typedef __bf16 bf16;
typedef __attribute__((ext_vector_type(16))) __bf16 v16bf;
typedef __attribute__((ext_vector_type(8)))  float  v8f;

static constexpr int BM = 128;   // block tile M
static constexpr int BN = 64;    // block tile N
static constexpr int BK = 32;    // block tile K (one bf16 WMMA k-step)
static constexpr int PAD = 8;    // LDS row padding (keeps 16B alignment, skews banks)
static constexpr int NT = 256;   // 8 waves

__device__ __forceinline__ bf16 f2bf(float x) { return (bf16)x; }

// ---- CDNA5 async global->LDS copy (16B per lane, tracked with ASYNCcnt) ----
__device__ __forceinline__ void async_copy_b128(unsigned lds_addr, const bf16* gptr) {
  asm volatile("global_load_async_to_lds_b128 %0, %1, off"
               :: "v"(lds_addr), "v"(gptr)
               : "memory");
}
__device__ __forceinline__ void wait_async0() {
  asm volatile("s_wait_asynccnt 0x0" ::: "memory");
}
__device__ __forceinline__ unsigned lds_addr_of(const void* p) {
  // generic shared-aperture address: low 32 bits are the LDS byte offset
  return (unsigned)(uintptr_t)p;
}

// ---------------- lambda / mask prep ----------------
__global__ void tpm_prep_lambda(const float* __restrict__ conf,
                                float* __restrict__ lam_m,
                                float* __restrict__ inv_cnt, int n) {
  __shared__ float ssum[NT];
  float cnt = 0.f;
  for (int b = threadIdx.x; b < n; b += NT) {
    float lam = 1.f - conf[b];
    lam = fminf(fmaxf(lam, 0.f), 1.f);
    float m = (lam > 0.3f) ? 1.f : 0.f;
    lam_m[b] = lam * m;
    cnt += m;
  }
  ssum[threadIdx.x] = cnt;
  __syncthreads();
  for (int s = NT / 2; s > 0; s >>= 1) {
    if (threadIdx.x < s) ssum[threadIdx.x] += ssum[threadIdx.x + s];
    __syncthreads();
  }
  if (threadIdx.x == 0) inv_cnt[0] = 1.f / (ssum[0] + 1e-6f);
}

// ---------------- fp32 -> bf16 converters ----------------
__global__ void tpm_cvt(const float* __restrict__ src, bf16* __restrict__ dst,
                        int n, float scale) {
  int i = blockIdx.x * blockDim.x + threadIdx.x;
  if (i < n) dst[i] = f2bf(src[i] * scale);
}

__global__ void tpm_cvt_rowscale(const float* __restrict__ src,
                                 const float* __restrict__ rowscale,
                                 bf16* __restrict__ dst, int n, int rowshift) {
  int i = blockIdx.x * blockDim.x + threadIdx.x;
  if (i < n) dst[i] = f2bf(src[i] * rowscale[i >> rowshift]);
}

// ---------------- fragment helpers (documented CDNA5 VGPR layouts) ----------------
union FragBF { v16bf v; uint4 q[2]; };

// A fragment (16x32 MxK): lane 0-15 M=lane, K blocks {0-7,16-23}; lane 16-31 K blocks {8-15,24-31}
__device__ __forceinline__ void load_frag_a(FragBF& f, const bf16* row, int hi) {
  f.q[0] = *(const uint4*)(row + hi * 8);
  f.q[1] = *(const uint4*)(row + 16 + hi * 8);
}
// B fragment (32x16 KxN), stored in LDS as [n][k]: lane 0-15 K blocks {0-7,16-23}; lane 16-31 K blocks {8-15,24-31}
__device__ __forceinline__ void load_frag_b(FragBF& f, const bf16* col, int hi) {
  f.q[0] = *(const uint4*)(col + hi * 8);
  f.q[1] = *(const uint4*)(col + 16 + hi * 8);
}

#define WMMA_BF16(ACC, Afr, Bfr) \
  (ACC) = __builtin_amdgcn_wmma_f32_16x16x32_bf16(false, (Afr).v, false, (Bfr).v, (short)0, (ACC), false, false)

// ---------------- generic C = A(MxK) @ B(KxN), bf16 in, fp32 or bf16 out ----------------
template <bool OUT_BF16>
__global__ __launch_bounds__(NT) void tpm_gemm_bf16(
    const bf16* __restrict__ A, const bf16* __restrict__ Bm, void* __restrict__ Cout,
    int K, int lda, int ldb, int ldc) {
  __shared__ alignas(16) bf16 As[BM][BK + PAD];
  __shared__ alignas(16) bf16 Bs[BN][BK + PAD];

  const int tid = threadIdx.x;
  const int wid = tid >> 5;
  const int lane = tid & 31;
  const int hi = (lane >= 16) ? 1 : 0;
  const int lr = lane & 15;
  const int wm = wid & 3;   // wave row: 32 rows each
  const int wn = wid >> 2;  // wave col: 32 cols each
  const int m0 = blockIdx.y * BM;
  const int n0 = blockIdx.x * BN;

  const unsigned asbase = lds_addr_of(&As[0][0]);

  v8f acc[2][2];
#pragma unroll
  for (int i = 0; i < 2; ++i)
#pragma unroll
    for (int j = 0; j < 2; ++j)
      acc[i][j] = v8f{0.f, 0.f, 0.f, 0.f, 0.f, 0.f, 0.f, 0.f};

  for (int k0 = 0; k0 < K; k0 += BK) {
    // A tile: row-major copy, async global->LDS (b128 = 8 bf16 per lane)
#pragma unroll
    for (int c = tid; c < BM * (BK / 8); c += NT) {
      int m = c / (BK / 8);
      int kv = (c % (BK / 8)) * 8;
      unsigned laddr = asbase + (unsigned)(m * (BK + PAD) + kv) * 2u;
      async_copy_b128(laddr, &A[(size_t)(m0 + m) * lda + k0 + kv]);
    }
    // B tile: transpose into Bs[n][k] so fragments are K-contiguous (b128 global reads)
    for (int i = tid; i < BK * (BN / 8); i += NT) {
      int k = i / (BN / 8);
      int nv = (i % (BN / 8)) * 8;
      uint4 d = *(const uint4*)&Bm[(size_t)(k0 + k) * ldb + n0 + nv];
      const bf16* p = (const bf16*)&d;
#pragma unroll
      for (int j = 0; j < 8; ++j) Bs[nv + j][k] = p[j];
    }
    wait_async0();
    __syncthreads();

    FragBF a[2], b[2];
#pragma unroll
    for (int i = 0; i < 2; ++i) load_frag_a(a[i], &As[wm * 32 + i * 16 + lr][0], hi);
#pragma unroll
    for (int j = 0; j < 2; ++j) load_frag_b(b[j], &Bs[wn * 32 + j * 16 + lr][0], hi);
#pragma unroll
    for (int i = 0; i < 2; ++i)
#pragma unroll
      for (int j = 0; j < 2; ++j) WMMA_BF16(acc[i][j], a[i], b[j]);
    __syncthreads();
  }

  // C/D layout: lane holds col = lane&15; VGPR v holds row v + 8*hi within 16-block
#pragma unroll
  for (int i = 0; i < 2; ++i)
#pragma unroll
    for (int j = 0; j < 2; ++j) {
      int col = n0 + wn * 32 + j * 16 + lr;
#pragma unroll
      for (int v = 0; v < 8; ++v) {
        int row = m0 + wm * 32 + i * 16 + hi * 8 + v;
        float val = acc[i][j][v];
        if (OUT_BF16)
          ((bf16*)Cout)[(size_t)row * ldc + col] = f2bf(val);
        else
          ((float*)Cout)[(size_t)row * ldc + col] = val;
      }
    }
}

// ---------------- memory update: out[h] = mem[h] + alpha * KW_h^T @ VW_h ----------------
// KW, VW are (Bdim x H*D) row-major bf16; per-head column slice h*D.
__global__ __launch_bounds__(NT) void tpm_delta_gemm(
    const bf16* __restrict__ KW, const bf16* __restrict__ VW,
    const float* __restrict__ memF, const float* __restrict__ inv_cnt,
    float* __restrict__ out, int D, int Bdim, int ldab) {
  __shared__ alignas(16) bf16 As[BM][BK + PAD];
  __shared__ alignas(16) bf16 Bs[BN][BK + PAD];

  const int tid = threadIdx.x;
  const int wid = tid >> 5;
  const int lane = tid & 31;
  const int hi = (lane >= 16) ? 1 : 0;
  const int lr = lane & 15;
  const int wm = wid & 3;
  const int wn = wid >> 2;
  const int m0 = blockIdx.y * BM;
  const int n0 = blockIdx.x * BN;
  const int h = blockIdx.z;
  const size_t hoff = (size_t)h * D;
  const size_t hbase = (size_t)h * D * D;

  v8f acc[2][2];
#pragma unroll
  for (int i = 0; i < 2; ++i)
#pragma unroll
    for (int j = 0; j < 2; ++j)
      acc[i][j] = v8f{0.f, 0.f, 0.f, 0.f, 0.f, 0.f, 0.f, 0.f};

  for (int k0 = 0; k0 < Bdim; k0 += BK) {
    // As[m][k] = KW[(k0+k), hoff + m0 + m]   (A = KW_h^T, contiguous in m; b128 reads)
    for (int i = tid; i < BK * (BM / 8); i += NT) {
      int k = i / (BM / 8);
      int mv = (i % (BM / 8)) * 8;
      uint4 d = *(const uint4*)&KW[(size_t)(k0 + k) * ldab + hoff + m0 + mv];
      const bf16* p = (const bf16*)&d;
#pragma unroll
      for (int j = 0; j < 8; ++j) As[mv + j][k] = p[j];
    }
    // Bs[n][k] = VW[(k0+k), hoff + n0 + n]   (b128 reads)
    for (int i = tid; i < BK * (BN / 8); i += NT) {
      int k = i / (BN / 8);
      int nv = (i % (BN / 8)) * 8;
      uint4 d = *(const uint4*)&VW[(size_t)(k0 + k) * ldab + hoff + n0 + nv];
      const bf16* p = (const bf16*)&d;
#pragma unroll
      for (int j = 0; j < 8; ++j) Bs[nv + j][k] = p[j];
    }
    __syncthreads();

    FragBF a[2], b[2];
#pragma unroll
    for (int i = 0; i < 2; ++i) load_frag_a(a[i], &As[wm * 32 + i * 16 + lr][0], hi);
#pragma unroll
    for (int j = 0; j < 2; ++j) load_frag_b(b[j], &Bs[wn * 32 + j * 16 + lr][0], hi);
#pragma unroll
    for (int i = 0; i < 2; ++i)
#pragma unroll
      for (int j = 0; j < 2; ++j) WMMA_BF16(acc[i][j], a[i], b[j]);
    __syncthreads();
  }

  const float alpha = inv_cnt[0];
#pragma unroll
  for (int i = 0; i < 2; ++i)
#pragma unroll
    for (int j = 0; j < 2; ++j) {
      int col = n0 + wn * 32 + j * 16 + lr;
#pragma unroll
      for (int v = 0; v < 8; ++v) {
        int row = m0 + wm * 32 + i * 16 + hi * 8 + v;
        size_t idx = hbase + (size_t)row * D + col;
        out[idx] = memF[idx] + alpha * acc[i][j][v];
      }
    }
}

extern "C" void kernel_launch(void* const* d_in, const int* in_sizes, int n_in,
                              void* d_out, int out_size, void* d_ws, size_t ws_size,
                              hipStream_t stream) {
  (void)in_sizes; (void)n_in; (void)out_size; (void)ws_size;
  const int Bt = 4096, D = 512, H = 16;
  const int HD = H * D;  // 8192

  const float* zq   = (const float*)d_in[0];
  const float* zs   = (const float*)d_in[1];
  const float* zv   = (const float*)d_in[2];
  const float* conf = (const float*)d_in[3];
  const float* Wk   = (const float*)d_in[4];
  const float* Wv   = (const float*)d_in[5];
  const float* Wo   = (const float*)d_in[6];
  const float* mem  = (const float*)d_in[7];

  char* ws = (char*)d_ws;
  size_t off = 0;
  auto carve = [&](size_t bytes) -> void* {
    void* p = ws + off;
    off = (off + bytes + 255) & ~(size_t)255;
    return p;
  };
  float* lam_m  = (float*)carve((size_t)Bt * 4);
  float* invcnt = (float*)carve(4);
  bf16* zq_b  = (bf16*)carve((size_t)Bt * D * 2);
  bf16* zs_b  = (bf16*)carve((size_t)Bt * D * 2);
  bf16* zv_b  = (bf16*)carve((size_t)Bt * D * 2);
  bf16* Wk_b  = (bf16*)carve((size_t)D * HD * 2);
  bf16* Wv_b  = (bf16*)carve((size_t)D * HD * 2);
  bf16* Wo_b  = (bf16*)carve((size_t)D * D * 2);
  bf16* mem_b = (bf16*)carve((size_t)HD * D * 2);
  bf16* Mp_b  = (bf16*)carve((size_t)HD * D * 2);   // (M_h @ Wo/H) flattened (8192 x 512)
  bf16* Kq_b  = (bf16*)carve((size_t)Bt * HD * 2);  // zq @ Wk
  bf16* KW_b  = (bf16*)carve((size_t)Bt * HD * 2);  // (lam_m * zs) @ Wk
  bf16* VW_b  = (bf16*)carve((size_t)Bt * HD * 2);  // zv @ Wv

  // 1) lambda / mask scalar prep
  tpm_prep_lambda<<<1, NT, 0, stream>>>(conf, lam_m, invcnt, Bt);

  // 2) conversions (fold 1/H into Wo, lam_m row-scale into zs)
  auto cgrid = [](int n) { return dim3((n + NT - 1) / NT); };
  tpm_cvt<<<cgrid(Bt * D), NT, 0, stream>>>(zq, zq_b, Bt * D, 1.f);
  tpm_cvt_rowscale<<<cgrid(Bt * D), NT, 0, stream>>>(zs, lam_m, zs_b, Bt * D, 9);
  tpm_cvt<<<cgrid(Bt * D), NT, 0, stream>>>(zv, zv_b, Bt * D, 1.f);
  tpm_cvt<<<cgrid(D * HD), NT, 0, stream>>>(Wk, Wk_b, D * HD, 1.f);
  tpm_cvt<<<cgrid(D * HD), NT, 0, stream>>>(Wv, Wv_b, D * HD, 1.f);
  tpm_cvt<<<cgrid(D * D), NT, 0, stream>>>(Wo, Wo_b, D * D, 1.f / (float)H);
  tpm_cvt<<<cgrid(HD * D), NT, 0, stream>>>(mem, mem_b, HD * D, 1.f);

  dim3 blk(NT);
  // 3) Mp = mem_flat(8192x512) @ (Wo/H)(512x512)  -> bf16
  tpm_gemm_bf16<true><<<dim3(D / BN, HD / BM), blk, 0, stream>>>(mem_b, Wo_b, Mp_b, D, D, D, D);
  // 4) projections: (4096x512) @ (512x8192) -> bf16
  tpm_gemm_bf16<true><<<dim3(HD / BN, Bt / BM), blk, 0, stream>>>(zq_b, Wk_b, Kq_b, D, D, HD, HD);
  tpm_gemm_bf16<true><<<dim3(HD / BN, Bt / BM), blk, 0, stream>>>(zs_b, Wk_b, KW_b, D, D, HD, HD);
  tpm_gemm_bf16<true><<<dim3(HD / BN, Bt / BM), blk, 0, stream>>>(zv_b, Wv_b, VW_b, D, D, HD, HD);
  // 5) v_ret = Kq(4096x8192) @ Mp(8192x512) -> fp32 d_out[0 : B*D)
  tpm_gemm_bf16<false><<<dim3(D / BN, Bt / BM), blk, 0, stream>>>(Kq_b, Mp_b, (float*)d_out, HD, HD, D, D);
  // 6) new_memory[h] = mem[h] + invcnt * KW_h^T @ VW_h -> fp32 d_out[B*D : B*D + H*D*D)
  tpm_delta_gemm<<<dim3(D / BN, D / BM, H), blk, 0, stream>>>(
      KW_b, VW_b, mem, invcnt, (float*)d_out + (size_t)Bt * D, D, Bt, HD);
}
